// SchNet_72602127171982
// MI455X (gfx1250) — matrix operand order repaired and checked by reading, not compiled, not run on hardware
//
#include <hip/hip_runtime.h>
#include <hip/hip_fp16.h>

typedef _Float16 half_t;
typedef __attribute__((ext_vector_type(16))) _Float16 v16h;
typedef __attribute__((ext_vector_type(8)))  float    v8f;

#define B_SZ   32
#define N_ATM  128
#define FDIM   128
#define NROWS  (B_SZ * N_ATM)            // 4096 atom rows
#define NPAIRB (B_SZ * N_ATM)            // 4096 (b,i) blocks
#define NG     300
#define NGPAD  320                        // 10 k-tiles of 32
#define WF1_HALVES (NGPAD * FDIM)         // 40960 halves = 80 KB

// shifted softplus: log(0.5*exp(x)+0.5)  — branch-free, fast-math ops only.
__device__ __forceinline__ float sspf(float x) {
  return __logf(fmaf(0.5f, __expf(x), 0.5f));
}

__device__ __forceinline__ v8f v8f_zero() {
  v8f z;
#pragma unroll
  for (int i = 0; i < 8; ++i) z[i] = 0.f;
  return z;
}

// A-fragment K index for 16-bit 16x32 A tile (ISA 7.12.2):
// lanes 0-15 hold K {0..7,16..23}, lanes 16-31 hold K {8..15,24..31}
__device__ __forceinline__ int a_kidx(int t, int lane) {
  return ((t >> 3) << 4) + ((lane >> 4) << 3) + (t & 7);
}

// ---------------------------------------------------------------------------
// Pack a [K x Ncols] row-major f32 weight into B-fragment order:
//   frag[(kt*nnt + nt)*32 + lane][t] = w[kt*32 + (t>>3)*16 + (lane>>4)*8 + (t&7)][nt*16 + (lane&15)]
// Zero-pads K up to Kpad.
// ---------------------------------------------------------------------------
__global__ void pack_b_frags(const float* __restrict__ w, int K, int Kpad,
                             int Ncols, half_t* __restrict__ out) {
  int idx = blockIdx.x * blockDim.x + threadIdx.x;   // one thread per (kt,nt,lane)
  int nnt = Ncols >> 4;
  int total = (Kpad >> 5) * nnt * 32;
  if (idx >= total) return;
  int lane = idx & 31;
  int nt   = (idx >> 5) % nnt;
  int kt   = (idx >> 5) / nnt;
  int n    = nt * 16 + (lane & 15);
  half_t* o = out + (size_t)idx * 16;
#pragma unroll
  for (int t = 0; t < 16; ++t) {
    int k = kt * 32 + ((t >> 3) << 4) + ((lane >> 4) << 3) + (t & 7);
    float v = (k < K) ? w[(size_t)k * Ncols + n] : 0.f;
    o[t] = (half_t)v;
  }
}

// ---------------------------------------------------------------------------
// x[row, h] = emb[z[row], h]
// ---------------------------------------------------------------------------
__global__ void init_x_kernel(const int* __restrict__ z,
                              const float* __restrict__ emb,
                              float* __restrict__ x) {
  int idx = blockIdx.x * blockDim.x + threadIdx.x;   // NROWS*FDIM threads
  int row = idx >> 7;
  int h   = idx & 127;
  x[idx] = emb[(size_t)z[row] * FDIM + h];
}

// ---------------------------------------------------------------------------
// Filter network, fused: per block = one (b,i), rows j = 0..127.
//   d[j] -> rbf built directly as WMMA A fragments (no rbf tensor in memory)
//   W1 = ssp(rbf @ w_f1 + b_f1) ;  W = W1 @ w_f2 + b_f2  -> f16 [b,i,j,h]
// w_f1 fragment bank (80 KB) is staged in LDS so per-WMMA B reads are
// short-latency ds_load_b128 instead of VMEM round trips.
// ---------------------------------------------------------------------------
__global__ __launch_bounds__(256) void filter_kernel(
    const float*  __restrict__ r,       // [B,N,3]
    const half_t* __restrict__ wf1f,    // frags, Kpad=320, N=128 (global)
    const half_t* __restrict__ wf2f,    // frags, K=128,  N=128 (global)
    const float*  __restrict__ b_f1,
    const float*  __restrict__ b_f2,
    half_t*       __restrict__ W) {     // [B,N,N,128] f16
  __shared__ half_t wf1s[WF1_HALVES];   // 80 KB staged w_f1 fragments
  __shared__ half_t a2[N_ATM * FDIM];   // 32 KB restage buffer
  __shared__ float  d_sh[N_ATM];

  int blk = blockIdx.x;                 // 0..4095 = (b,i)
  int b   = blk >> 7;
  int i   = blk & 127;
  int tid  = threadIdx.x;
  int wave = tid >> 5;
  int lane = tid & 31;

  // stage w_f1 fragments: 5120 uint4, 20 per thread, fully coalesced.
  {
    const uint4* src = (const uint4*)wf1f;
    uint4* dst = (uint4*)wf1s;
#pragma unroll
    for (int k = 0; k < 20; ++k) dst[tid + k * 256] = src[tid + k * 256];
  }

  if (tid < N_ATM) {
    const float* ri = r + (size_t)(b * N_ATM + i)   * 3;
    const float* rj = r + (size_t)(b * N_ATM + tid) * 3;
    float dx = ri[0] - rj[0], dy = ri[1] - rj[1], dz = ri[2] - rj[2];
    d_sh[tid] = sqrtf(dx * dx + dy * dy + dz * dz + 1e-12f);
  }

  // hoist bias loads so they are in flight during the MMA loops
  float bias1 = b_f1[(lane & 15)];      // per-nt offset added later
  float bias1v[8];
#pragma unroll
  for (int nt = 0; nt < 8; ++nt) bias1v[nt] = b_f1[nt * 16 + (lane & 15)];
  float bias2v[8];
#pragma unroll
  for (int nt = 0; nt < 8; ++nt) bias2v[nt] = b_f2[nt * 16 + (lane & 15)];
  (void)bias1;

  __syncthreads();

  int   mrow = wave * 16 + (lane & 15);     // A-matrix row (pair row j)
  float dm   = d_sh[mrow];

  // ---- GEMM 1: rbf(320) x w_f1 -> 16x128 per wave ----
  v8f acc[8];
#pragma unroll
  for (int nt = 0; nt < 8; ++nt) acc[nt] = v8f_zero();

  const float cstep = 30.0f / 299.0f;       // linspace(0,30,300) step
  for (int kt = 0; kt < 10; ++kt) {
    v16h a;
#pragma unroll
    for (int t = 0; t < 16; ++t) {
      int   k = kt * 32 + a_kidx(t, lane);
      float e = dm - (float)k * cstep;
      float v = (k < NG) ? __expf(-10.0f * e * e) : 0.f;
      a[t] = (half_t)v;
    }
#pragma unroll
    for (int nt = 0; nt < 8; ++nt) {
      v16h bfr = *(const v16h*)(wf1s + (size_t)((kt * 8 + nt) * 32 + lane) * 16);
      acc[nt] = __builtin_amdgcn_wmma_f32_16x16x32_f16(
          false, a, false, bfr, (short)0, acc[nt], false, false);
    }
  }

  // epilogue 1: + b_f1, ssp, restage as f16 [row][h] in LDS
#pragma unroll
  for (int nt = 0; nt < 8; ++nt) {
#pragma unroll
    for (int v = 0; v < 8; ++v) {
      int rrow = wave * 16 + ((lane >> 4) << 3) + v;
      a2[rrow * FDIM + nt * 16 + (lane & 15)] =
          (half_t)sspf(acc[nt][v] + bias1v[nt]);
    }
  }
  __syncthreads();

  // ---- GEMM 2: a2(128) x w_f2 ----
  v8f acc2[8];
#pragma unroll
  for (int nt = 0; nt < 8; ++nt) acc2[nt] = v8f_zero();

  const half_t* rowp = &a2[mrow * FDIM];
  for (int kt = 0; kt < 4; ++kt) {
    v16h a;
    int base = kt * 32 + ((lane >> 4) << 3);
#pragma unroll
    for (int t = 0; t < 8; ++t) {
      a[t]     = rowp[base + t];
      a[8 + t] = rowp[base + 16 + t];
    }
#pragma unroll
    for (int nt = 0; nt < 8; ++nt) {
      v16h bfr = *(const v16h*)(wf2f + (size_t)((kt * 8 + nt) * 32 + lane) * 16);
      acc2[nt] = __builtin_amdgcn_wmma_f32_16x16x32_f16(
          false, a, false, bfr, (short)0, acc2[nt], false, false);
    }
  }

  // restage result through LDS, then fully-coalesced b128 stores of W
  __syncthreads();   // all a2 reads done
#pragma unroll
  for (int nt = 0; nt < 8; ++nt) {
#pragma unroll
    for (int v = 0; v < 8; ++v) {
      int rrow = wave * 16 + ((lane >> 4) << 3) + v;
      a2[rrow * FDIM + nt * 16 + (lane & 15)] =
          (half_t)(acc2[nt][v] + bias2v[nt]);
    }
  }
  __syncthreads();
  uint4* w4 = (uint4*)(W + (size_t)blk * N_ATM * FDIM);
  const uint4* a4 = (const uint4*)a2;
#pragma unroll
  for (int k = 0; k < 8; ++k) w4[tid + k * 256] = a4[tid + k * 256];
}

// ---------------------------------------------------------------------------
// f = x @ w_in2f  (no bias), f stored plain f16 [row][h]
// ---------------------------------------------------------------------------
__global__ __launch_bounds__(256) void gemm_xf_kernel(
    const float* __restrict__ x, const half_t* __restrict__ wfrag,
    half_t* __restrict__ f) {
  int row0 = blockIdx.x * 128;
  int tid = threadIdx.x, wave = tid >> 5, lane = tid & 31;
  const float* arow = x + (size_t)(row0 + wave * 16 + (lane & 15)) * FDIM;

  v8f acc[8];
#pragma unroll
  for (int nt = 0; nt < 8; ++nt) acc[nt] = v8f_zero();

  for (int kt = 0; kt < 4; ++kt) {
    v16h a;
    int base = kt * 32 + ((lane >> 4) << 3);
#pragma unroll
    for (int t = 0; t < 8; ++t) {
      a[t]     = (half_t)arow[base + t];
      a[8 + t] = (half_t)arow[base + 16 + t];
    }
#pragma unroll
    for (int nt = 0; nt < 8; ++nt) {
      v16h bfr = *(const v16h*)(wfrag + (size_t)((kt * 8 + nt) * 32 + lane) * 16);
      acc[nt] = __builtin_amdgcn_wmma_f32_16x16x32_f16(
          false, a, false, bfr, (short)0, acc[nt], false, false);
    }
  }
#pragma unroll
  for (int nt = 0; nt < 8; ++nt)
#pragma unroll
    for (int v = 0; v < 8; ++v) {
      int rrow = row0 + wave * 16 + ((lane >> 4) << 3) + v;
      f[(size_t)rrow * FDIM + nt * 16 + (lane & 15)] = (half_t)acc[nt][v];
    }
}

// ---------------------------------------------------------------------------
// cfconv: y[b,i,h] = sum_j W[b,i,j,h] * f[b,j,h]   (bandwidth bound: streams W)
// block = (b,i), 128 threads. Thread = (j_sub 0..7, h-group of 8) so every W
// read is a coalesced b128; partial sums over j_sub reduced through LDS.
// ---------------------------------------------------------------------------
__global__ __launch_bounds__(128) void cfconv_kernel(
    const half_t* __restrict__ W, const half_t* __restrict__ f,
    float* __restrict__ y) {
  __shared__ half_t fl[N_ATM * FDIM];   // 32 KB: f[b,:,:]
  __shared__ float  red[128 * 8];       // 4 KB partial sums
  int blk = blockIdx.x;
  int b   = blk >> 7;
  int tid = threadIdx.x;

  const uint4* fb4 = (const uint4*)(f + (size_t)b * N_ATM * FDIM);
  uint4* fl4 = (uint4*)fl;
#pragma unroll
  for (int k = 0; k < 16; ++k) fl4[tid + k * 128] = fb4[tid + k * 128];
  __syncthreads();

  int jsub = tid >> 4;          // 0..7
  int h0   = (tid & 15) * 8;    // 8 h-columns per thread
  const half_t* Wb = W + (size_t)blk * N_ATM * FDIM;

  float acc[8];
#pragma unroll
  for (int t = 0; t < 8; ++t) acc[t] = 0.f;

#pragma unroll 4
  for (int j = jsub; j < N_ATM; j += 8) {
    uint4 wv = *(const uint4*)(Wb + j * FDIM + h0);   // 8 halves of W
    uint4 fv = *(const uint4*)(fl + j * FDIM + h0);   // 8 halves of f
    const half_t* wp = (const half_t*)&wv;
    const half_t* fp = (const half_t*)&fv;
#pragma unroll
    for (int t = 0; t < 8; ++t)
      acc[t] = fmaf((float)wp[t], (float)fp[t], acc[t]);
  }
  float* myred = red + tid * 8;
#pragma unroll
  for (int t = 0; t < 8; ++t) myred[t] = acc[t];
  __syncthreads();

  // reduce the 8 j_sub partials; 128 threads, one (h-group, t) each
  int hgrp = tid >> 3, t = tid & 7;
  float s = 0.f;
#pragma unroll
  for (int js = 0; js < 8; ++js) s += red[(js * 16 + hgrp) * 8 + t];
  y[(size_t)blk * FDIM + hgrp * 8 + t] = s;
}

// ---------------------------------------------------------------------------
// interaction epilogue: x += (ssp(y @ w_f2out + b_f2out)) @ w_out + b_out
// ---------------------------------------------------------------------------
__global__ __launch_bounds__(256) void update_kernel(
    const float* __restrict__ y, const half_t* __restrict__ wf2outf,
    const float* __restrict__ b_f2out, const half_t* __restrict__ woutf,
    const float* __restrict__ b_out, float* __restrict__ x) {
  __shared__ half_t a2[128 * FDIM];
  int row0 = blockIdx.x * 128;
  int tid = threadIdx.x, wave = tid >> 5, lane = tid & 31;
  int mloc = wave * 16 + (lane & 15);
  const float* arow = y + (size_t)(row0 + mloc) * FDIM;

  float biasAv[8], biasBv[8];
#pragma unroll
  for (int nt = 0; nt < 8; ++nt) {
    biasAv[nt] = b_f2out[nt * 16 + (lane & 15)];
    biasBv[nt] = b_out[nt * 16 + (lane & 15)];
  }

  v8f acc[8];
#pragma unroll
  for (int nt = 0; nt < 8; ++nt) acc[nt] = v8f_zero();
  for (int kt = 0; kt < 4; ++kt) {
    v16h a;
    int base = kt * 32 + ((lane >> 4) << 3);
#pragma unroll
    for (int t = 0; t < 8; ++t) {
      a[t]     = (half_t)arow[base + t];
      a[8 + t] = (half_t)arow[base + 16 + t];
    }
#pragma unroll
    for (int nt = 0; nt < 8; ++nt) {
      v16h bfr = *(const v16h*)(wf2outf + (size_t)((kt * 8 + nt) * 32 + lane) * 16);
      acc[nt] = __builtin_amdgcn_wmma_f32_16x16x32_f16(
          false, a, false, bfr, (short)0, acc[nt], false, false);
    }
  }
#pragma unroll
  for (int nt = 0; nt < 8; ++nt) {
#pragma unroll
    for (int v = 0; v < 8; ++v) {
      int rrow = wave * 16 + ((lane >> 4) << 3) + v;
      a2[rrow * FDIM + nt * 16 + (lane & 15)] =
          (half_t)sspf(acc[nt][v] + biasAv[nt]);
    }
  }
  __syncthreads();

  v8f acc2[8];
#pragma unroll
  for (int nt = 0; nt < 8; ++nt) acc2[nt] = v8f_zero();
  const half_t* rowp = &a2[mloc * FDIM];
  for (int kt = 0; kt < 4; ++kt) {
    v16h a;
    int base = kt * 32 + ((lane >> 4) << 3);
#pragma unroll
    for (int t = 0; t < 8; ++t) {
      a[t]     = rowp[base + t];
      a[8 + t] = rowp[base + 16 + t];
    }
#pragma unroll
    for (int nt = 0; nt < 8; ++nt) {
      v16h bfr = *(const v16h*)(woutf + (size_t)((kt * 8 + nt) * 32 + lane) * 16);
      acc2[nt] = __builtin_amdgcn_wmma_f32_16x16x32_f16(
          false, a, false, bfr, (short)0, acc2[nt], false, false);
    }
  }
#pragma unroll
  for (int nt = 0; nt < 8; ++nt) {
#pragma unroll
    for (int v = 0; v < 8; ++v) {
      int rrow = row0 + wave * 16 + ((lane >> 4) << 3) + v;
      x[(size_t)rrow * FDIM + nt * 16 + (lane & 15)] += acc2[nt][v] + biasBv[nt];
    }
  }
}

// ---------------------------------------------------------------------------
// readout: out[row] = ssp(x @ w_aw1 + b_aw1) . w_aw2 + b_aw2
// ---------------------------------------------------------------------------
__global__ __launch_bounds__(256) void readout_kernel(
    const float* __restrict__ x, const half_t* __restrict__ waw1f,
    const float* __restrict__ b_aw1, const float* __restrict__ w_aw2,
    const float* __restrict__ b_aw2, float* __restrict__ out) {
  __shared__ half_t u[128 * FDIM];
  int row0 = blockIdx.x * 128;
  int tid = threadIdx.x, wave = tid >> 5, lane = tid & 31;
  const float* arow = x + (size_t)(row0 + wave * 16 + (lane & 15)) * FDIM;

  float biasv[8];
#pragma unroll
  for (int nt = 0; nt < 8; ++nt) biasv[nt] = b_aw1[nt * 16 + (lane & 15)];

  v8f acc[8];
#pragma unroll
  for (int nt = 0; nt < 8; ++nt) acc[nt] = v8f_zero();
  for (int kt = 0; kt < 4; ++kt) {
    v16h a;
    int base = kt * 32 + ((lane >> 4) << 3);
#pragma unroll
    for (int t = 0; t < 8; ++t) {
      a[t]     = (half_t)arow[base + t];
      a[8 + t] = (half_t)arow[base + 16 + t];
    }
#pragma unroll
    for (int nt = 0; nt < 8; ++nt) {
      v16h bfr = *(const v16h*)(waw1f + (size_t)((kt * 8 + nt) * 32 + lane) * 16);
      acc[nt] = __builtin_amdgcn_wmma_f32_16x16x32_f16(
          false, a, false, bfr, (short)0, acc[nt], false, false);
    }
  }
#pragma unroll
  for (int nt = 0; nt < 8; ++nt) {
#pragma unroll
    for (int v = 0; v < 8; ++v) {
      int rrow = wave * 16 + ((lane >> 4) << 3) + v;
      u[rrow * FDIM + nt * 16 + (lane & 15)] =
          (half_t)sspf(acc[nt][v] + biasv[nt]);
    }
  }
  __syncthreads();

  if (tid < 128) {
    float s = b_aw2[0];
#pragma unroll 8
    for (int h = 0; h < FDIM; ++h)
      s = fmaf((float)u[tid * FDIM + h], w_aw2[h], s);
    out[row0 + tid] = s;
  }
}

// ---------------------------------------------------------------------------
extern "C" void kernel_launch(void* const* d_in, const int* in_sizes, int n_in,
                              void* d_out, int out_size, void* d_ws,
                              size_t ws_size, hipStream_t stream) {
  (void)in_sizes; (void)n_in; (void)out_size; (void)ws_size;

  const int*   z       = (const int*)  d_in[0];
  const float* r       = (const float*)d_in[1];
  const float* emb     = (const float*)d_in[2];
  const float* w_in2f  = (const float*)d_in[3];
  const float* w_f1    = (const float*)d_in[4];
  const float* b_f1    = (const float*)d_in[5];
  const float* w_f2    = (const float*)d_in[6];
  const float* b_f2    = (const float*)d_in[7];
  const float* w_f2out = (const float*)d_in[8];
  const float* b_f2out = (const float*)d_in[9];
  const float* w_out   = (const float*)d_in[10];
  const float* b_out   = (const float*)d_in[11];
  const float* w_aw1   = (const float*)d_in[12];
  const float* b_aw1   = (const float*)d_in[13];
  const float* w_aw2   = (const float*)d_in[14];
  const float* b_aw2   = (const float*)d_in[15];
  float* out = (float*)d_out;

  char* ws = (char*)d_ws;
  size_t off = 0;
  auto carve = [&](size_t bytes) -> void* {
    void* p = ws + off;
    off += (bytes + 255) & ~(size_t)255;
    return p;
  };
  half_t* W        = (half_t*)carve((size_t)NPAIRB * N_ATM * FDIM * 2);  // 128 MB
  half_t* wf1f     = (half_t*)carve((size_t)NGPAD * FDIM * 2);
  half_t* wf2f     = (half_t*)carve((size_t)FDIM * FDIM * 2);
  half_t* win2ff   = (half_t*)carve((size_t)FDIM * FDIM * 2);
  half_t* wf2outf  = (half_t*)carve((size_t)FDIM * FDIM * 2);
  half_t* woutf    = (half_t*)carve((size_t)FDIM * FDIM * 2);
  half_t* waw1f    = (half_t*)carve((size_t)FDIM * FDIM * 2);
  float*  x        = (float*) carve((size_t)NROWS * FDIM * 4);
  half_t* f        = (half_t*)carve((size_t)NROWS * FDIM * 2);
  float*  y        = (float*) carve((size_t)NROWS * FDIM * 4);

  // pack weights into B-fragment layout
  pack_b_frags<<<10, 256, 0, stream>>>(w_f1,    NG,   NGPAD, FDIM, wf1f);
  pack_b_frags<<<4,  256, 0, stream>>>(w_f2,    FDIM, FDIM,  FDIM, wf2f);
  pack_b_frags<<<4,  256, 0, stream>>>(w_in2f,  FDIM, FDIM,  FDIM, win2ff);
  pack_b_frags<<<4,  256, 0, stream>>>(w_f2out, FDIM, FDIM,  FDIM, wf2outf);
  pack_b_frags<<<4,  256, 0, stream>>>(w_out,   FDIM, FDIM,  FDIM, woutf);
  pack_b_frags<<<4,  256, 0, stream>>>(w_aw1,   FDIM, FDIM,  FDIM, waw1f);

  // x0 = emb[z]
  init_x_kernel<<<(NROWS * FDIM) / 256, 256, 0, stream>>>(z, emb, x);

  // filter network -> W (f16), computed once, shared across interactions
  filter_kernel<<<NPAIRB, 256, 0, stream>>>(r, wf1f, wf2f, b_f1, b_f2, W);

  // 3 weight-shared interaction blocks
  for (int it = 0; it < 3; ++it) {
    gemm_xf_kernel<<<NROWS / 128, 256, 0, stream>>>(x, win2ff, f);
    cfconv_kernel <<<NPAIRB, 128, 0, stream>>>(W, f, y);
    update_kernel <<<NROWS / 128, 256, 0, stream>>>(y, wf2outf, b_f2out,
                                                    woutf, b_out, x);
  }

  // readout
  readout_kernel<<<NROWS / 128, 256, 0, stream>>>(x, waw1f, b_aw1, w_aw2,
                                                  b_aw2, out);
}
